// Block_46523085750957
// MI455X (gfx1250) — compile-verified, matrix operand order checked by reading, not run on hardware
//
#include <hip/hip_runtime.h>

// ---------------------------------------------------------------------------
// MI455X / gfx1250 transformer block. wave32, WMMA bf16 (f32 accumulate).
// Fragment layout (CDNA5 ISA):
//   A 16x32 bf16: lane L -> row = L%16, khalf = L/16;
//     elements e=0..7  hold K = kbase + 8*khalf + e        (16 contiguous bytes)
//     elements e=8..15 hold K = kbase + 16 + 8*khalf + e-8 (16 contiguous bytes)
//   B 32x16 bf16: same e->K mapping, lane selects column N = L%16.
//     => feed B from a TRANSPOSED (N-major) buffer for contiguous loads.
//   C/D 16x16 f32: lane L -> col N = L%16; VGPR r -> row M = r + 8*(L/16).
// GEMMs use a 32x64 wave tile with double-buffered fragments so global_load
// latency overlaps the 8 WMMAs of the previous k-step.
// ---------------------------------------------------------------------------

typedef __attribute__((ext_vector_type(16))) __bf16    v16bf;
typedef __attribute__((ext_vector_type(8)))  float     v8f;
typedef __attribute__((ext_vector_type(4)))  unsigned  u32x4;

union FragU { v16bf v; u32x4 q[2]; };

__device__ __forceinline__ unsigned short f2bf(float f) {
  unsigned u = __float_as_uint(f);
  unsigned r = u + 0x7FFFu + ((u >> 16) & 1u);   // round-to-nearest-even
  return (unsigned short)(r >> 16);
}

// p must point at (row base + kbase + 8*khalf); two b128 loads.
__device__ __forceinline__ v16bf ldfrag(const unsigned short* __restrict__ p) {
  FragU f;
  f.q[0] = *(const u32x4*)(p);        // K = kbase + 8*khalf + 0..7
  f.q[1] = *(const u32x4*)(p + 16);   // K = kbase + 16 + 8*khalf + 0..7
  return f.v;
}

__device__ __forceinline__ v16bf load_frag(const unsigned short* __restrict__ base,
                                           int ldk, int row, int kbase, int khalf) {
  return ldfrag(base + (size_t)row * ldk + kbase + (khalf << 3));
}

#define WMMA_BF16(a, b, c) \
  __builtin_amdgcn_wmma_f32_16x16x32_bf16(false, (a), false, (b), (short)0, (c), false, false)

// ------------------------- constants --------------------------------------
#define BATCH   64
#define SEQ     256
#define CEMB    512
#define NHEAD   8
#define HDIM    64
#define MROWS   (BATCH * SEQ)       // 16384
#define FF      (4 * CEMB)          // 2048
#define SCALE   0.044194173824159216f   // 512^-0.5

// ------------------------- weight conversion ------------------------------
__global__ void conv_qkvT_kernel(const float* __restrict__ Wk,
                                 const float* __restrict__ Wq,
                                 const float* __restrict__ Wv,
                                 unsigned short* __restrict__ out) {
  int idx = blockIdx.x * 256 + threadIdx.x;
  if (idx >= 3 * CEMB * CEMB) return;
  int j = idx >> 9, c = idx & 511;
  int jj = j & 511, h = jj >> 6, d = jj & 63;
  const float* src = (j < 512) ? Wk : (j < 1024) ? Wq : Wv;
  out[idx] = f2bf(src[((size_t)h * CEMB + c) * HDIM + d]);
}

// out[N][K] = bf16(in[K][N])
__global__ void convT_kernel(const float* __restrict__ in,
                             unsigned short* __restrict__ out, int K, int N) {
  int idx = blockIdx.x * 256 + threadIdx.x;
  if (idx >= N * K) return;
  int n = idx / K, k = idx - n * K;
  out[idx] = f2bf(in[(size_t)k * N + n]);
}

// ------------------------- layernorm -> bf16 ------------------------------
__global__ void ln_bf16_kernel(const float* __restrict__ x, const float* __restrict__ g,
                               const float* __restrict__ bb, unsigned short* __restrict__ out) {
  __shared__ float red[256];
  int row = blockIdx.x, tid = threadIdx.x;
  const float* xr = x + (size_t)row * CEMB;
  float a0 = xr[tid], a1 = xr[tid + 256];
  red[tid] = a0 + a1;
  __syncthreads();
  for (int s = 128; s > 0; s >>= 1) { if (tid < s) red[tid] += red[tid + s]; __syncthreads(); }
  float mean = red[0] * (1.0f / CEMB);
  __syncthreads();
  float d0 = a0 - mean, d1 = a1 - mean;
  red[tid] = d0 * d0 + d1 * d1;
  __syncthreads();
  for (int s = 128; s > 0; s >>= 1) { if (tid < s) red[tid] += red[tid + s]; __syncthreads(); }
  float rinv = rsqrtf(red[0] * (1.0f / CEMB) + 1e-5f);
  unsigned short* o = out + (size_t)row * CEMB;
  o[tid]       = f2bf(d0 * rinv * g[tid]       + bb[tid]);
  o[tid + 256] = f2bf(d1 * rinv * g[tid + 256] + bb[tid + 256]);
}

// ------------------------- GEMM core (32x64 wave tile, double-buffered) ----
// acc[mt][nt] accumulated over K; caller provides epilogue.
#define GEMM_CORE(A_, BT_, K_, tm_, tn_, lrow_, khalf_, acc_)                          \
  const unsigned short* pa0 = (A_) + (size_t)((tm_) * 32 + (lrow_)) * (K_) + ((khalf_) << 3); \
  const unsigned short* pa1 = pa0 + (size_t)16 * (K_);                                 \
  const unsigned short* pb0 = (BT_) + (size_t)((tn_) * 64 + (lrow_)) * (K_) + ((khalf_) << 3); \
  const unsigned short* pb1 = pb0 + (size_t)16 * (K_);                                 \
  const unsigned short* pb2 = pb0 + (size_t)32 * (K_);                                 \
  const unsigned short* pb3 = pb0 + (size_t)48 * (K_);                                 \
  v16bf a0 = ldfrag(pa0), a1 = ldfrag(pa1);                                            \
  v16bf b0 = ldfrag(pb0), b1 = ldfrag(pb1), b2 = ldfrag(pb2), b3 = ldfrag(pb3);        \
  for (int k = 32; k <= (K_); k += 32) {                                               \
    v16bf na0 = a0, na1 = a1, nb0 = b0, nb1 = b1, nb2 = b2, nb3 = b3;                  \
    if (k < (K_)) {                                                                    \
      na0 = ldfrag(pa0 + k); na1 = ldfrag(pa1 + k);                                    \
      nb0 = ldfrag(pb0 + k); nb1 = ldfrag(pb1 + k);                                    \
      nb2 = ldfrag(pb2 + k); nb3 = ldfrag(pb3 + k);                                    \
      if (k + 128 < (K_)) {            /* stream-prefetch the activation rows */       \
        __builtin_prefetch(pa0 + k + 128, 0, 1);                                       \
        __builtin_prefetch(pa1 + k + 128, 0, 1);                                       \
      }                                                                                \
    }                                                                                  \
    acc_[0][0] = WMMA_BF16(a0, b0, acc_[0][0]);                                        \
    acc_[0][1] = WMMA_BF16(a0, b1, acc_[0][1]);                                        \
    acc_[0][2] = WMMA_BF16(a0, b2, acc_[0][2]);                                        \
    acc_[0][3] = WMMA_BF16(a0, b3, acc_[0][3]);                                        \
    acc_[1][0] = WMMA_BF16(a1, b0, acc_[1][0]);                                        \
    acc_[1][1] = WMMA_BF16(a1, b1, acc_[1][1]);                                        \
    acc_[1][2] = WMMA_BF16(a1, b2, acc_[1][2]);                                        \
    acc_[1][3] = WMMA_BF16(a1, b3, acc_[1][3]);                                        \
    a0 = na0; a1 = na1; b0 = nb0; b1 = nb1; b2 = nb2; b3 = nb3;                        \
  }

// ------------------------- QKV GEMM ---------------------------------------
// C[16384][1536] = hln x WqkvT^T; scatter k,q -> (B,H,T,64), v -> vT (B,H,64,T).
__global__ __launch_bounds__(256) void gemm_qkv_kernel(
    const unsigned short* __restrict__ A, const unsigned short* __restrict__ BT,
    unsigned short* __restrict__ kbuf, unsigned short* __restrict__ qbuf,
    unsigned short* __restrict__ vT) {
  const int K = CEMB, tilesN = (3 * CEMB) / 64;  // 24
  int w = blockIdx.x * 8 + (threadIdx.x >> 5);
  int lane = threadIdx.x & 31, lrow = lane & 15, khalf = lane >> 4;
  int tm = w / tilesN, tn = w - tm * tilesN;
  v8f acc[2][4] = {};
  GEMM_CORE(A, BT, K, tm, tn, lrow, khalf, acc)
#pragma unroll
  for (int mt = 0; mt < 2; mt++) {
#pragma unroll
    for (int nt = 0; nt < 4; nt++) {
      int j = tn * 64 + nt * 16 + lrow;
      int jj = j & 511, h = jj >> 6, d = jj & 63;
#pragma unroll
      for (int r = 0; r < 8; r++) {
        int m = tm * 32 + mt * 16 + r + khalf * 8;
        int b = m >> 8, t = m & 255, bh = b * NHEAD + h;
        unsigned short val = f2bf(acc[mt][nt][r]);
        if (j < 512)       kbuf[((size_t)bh * SEQ + t) * HDIM + d] = val;
        else if (j < 1024) qbuf[((size_t)bh * SEQ + t) * HDIM + d] = val;
        else               vT[((size_t)bh * HDIM + d) * SEQ + t]   = val;
      }
    }
  }
}

// ------------------------- attention --------------------------------------
// One workgroup per (b,h, 32-row block). S = k @ q^T * scale (reference swaps
// q/k), causal mask, softmax over s, att = P @ v. LDS ~49KB.
__global__ __launch_bounds__(256) void attn_kernel(
    const unsigned short* __restrict__ kbuf, const unsigned short* __restrict__ qbuf,
    const unsigned short* __restrict__ vT, unsigned short* __restrict__ attbf) {
  extern __shared__ unsigned char smem[];
  float*          S   = (float*)smem;                                  // [32][256] f32
  unsigned short* P   = (unsigned short*)(smem + 32 * 256 * 4);        // [32][256] bf16
  float*          red = (float*)(smem + 32 * 256 * 4 + 32 * 256 * 2);  // [32][8]

  int bh = blockIdx.x >> 3, blkrow = blockIdx.x & 7;
  int t0 = blkrow * 32;
  int b = bh >> 3, h = bh & 7;
  const unsigned short* kb = kbuf + (size_t)bh * SEQ * HDIM;
  const unsigned short* qb = qbuf + (size_t)bh * SEQ * HDIM;
  const unsigned short* vb = vT   + (size_t)bh * HDIM * SEQ;
  int tid = threadIdx.x, lane = tid & 31, wave = tid >> 5;
  int lrow = lane & 15, khalf = lane >> 4;

  // ---- phase 1: scores (32 tiles of 16x16; 4 per wave; skip fully-masked)
#pragma unroll
  for (int i = 0; i < 4; i++) {
    int idx = wave * 4 + i;
    int rt = idx >> 4, ct = idx & 15;
    int scol = ct * 16 + lrow;
    if (ct * 16 > t0 + 31) {               // entire tile above the diagonal
#pragma unroll
      for (int r = 0; r < 8; r++) S[(rt * 16 + r + khalf * 8) * 256 + scol] = -1e30f;
      continue;
    }
    v8f acc = {};
#pragma unroll
    for (int k = 0; k < HDIM; k += 32) {
      v16bf a  = load_frag(kb, HDIM, t0 + rt * 16 + lrow, k, khalf);  // rows of k
      v16bf bq = load_frag(qb, HDIM, scol, k, khalf);                 // q as B^T
      acc = WMMA_BF16(a, bq, acc);
    }
#pragma unroll
    for (int r = 0; r < 8; r++) {
      int tl = rt * 16 + r + khalf * 8;
      int tg = t0 + tl;
      S[tl * 256 + scol] = (scol <= tg) ? acc[r] * SCALE : -1e30f;
    }
  }
  __syncthreads();

  // ---- phase 2: row softmax (row r handled by 8 threads, 32 elems each)
  {
    int r = tid >> 3, ch = tid & 7;
    float* Sr = S + r * 256;
    float lmax = -1e30f;
    for (int s = ch * 32; s < ch * 32 + 32; s++) lmax = fmaxf(lmax, Sr[s]);
    red[r * 8 + ch] = lmax;
    __syncthreads();
    float m = red[r * 8];
#pragma unroll
    for (int j = 1; j < 8; j++) m = fmaxf(m, red[r * 8 + j]);
    __syncthreads();
    float lsum = 0.f;
    for (int s = ch * 32; s < ch * 32 + 32; s++) {
      float e = __expf(Sr[s] - m);
      Sr[s] = e; lsum += e;
    }
    red[r * 8 + ch] = lsum;
    __syncthreads();
    float sum = 0.f;
#pragma unroll
    for (int j = 0; j < 8; j++) sum += red[r * 8 + j];
    float inv = 1.0f / sum;
    unsigned short* Pr = P + r * 256;
    for (int s = ch * 32; s < ch * 32 + 32; s++) Pr[s] = f2bf(Sr[s] * inv);
  }
  __syncthreads();

  // ---- phase 3: att = P @ V (8 tiles, one per wave); K truncated causally
  {
    int rt = wave >> 2, cd = wave & 3;
    v8f acc = {};
    int kmax = t0 + 32;                       // P is zero beyond the diagonal
    for (int s = 0; s < kmax; s += 32) {
      v16bf a  = load_frag(P,  256, rt * 16 + lrow, s, khalf);   // LDS b128 loads
      v16bf bv = load_frag(vb, 256, cd * 16 + lrow, s, khalf);   // vT rows = d
      acc = WMMA_BF16(a, bv, acc);
    }
#pragma unroll
    for (int r = 0; r < 8; r++) {
      int tl = rt * 16 + r + khalf * 8;
      int mrow = b * SEQ + t0 + tl;
      int col = h * HDIM + cd * 16 + lrow;
      attbf[(size_t)mrow * CEMB + col] = f2bf(acc[r]);
    }
  }
}

// ------------------------- generic GEMM + epilogue ------------------------
// EPI 0: outf = resid + acc + bias   (f32)   [out-proj, MLP2]
// EPI 1: outb = bf16(relu(acc+bias))         [MLP1]
template <int EPI>
__global__ __launch_bounds__(256) void gemm_bias_kernel(
    const unsigned short* __restrict__ A, const unsigned short* __restrict__ BT,
    int N, int K, const float* __restrict__ bias, const float* __restrict__ resid,
    float* __restrict__ outf, unsigned short* __restrict__ outb) {
  int tilesN = N >> 6;
  int w = blockIdx.x * 8 + (threadIdx.x >> 5);
  int lane = threadIdx.x & 31, lrow = lane & 15, khalf = lane >> 4;
  int tm = w / tilesN, tn = w - tm * tilesN;
  v8f acc[2][4] = {};
  GEMM_CORE(A, BT, K, tm, tn, lrow, khalf, acc)
#pragma unroll
  for (int mt = 0; mt < 2; mt++) {
#pragma unroll
    for (int nt = 0; nt < 4; nt++) {
      int n = tn * 64 + nt * 16 + lrow;
      float bn = bias[n];
#pragma unroll
      for (int r = 0; r < 8; r++) {
        int m = tm * 32 + mt * 16 + r + khalf * 8;
        float v = acc[mt][nt][r] + bn;
        if (EPI == 0) outf[(size_t)m * N + n] = resid[(size_t)m * N + n] + v;
        else          outb[(size_t)m * N + n] = f2bf(fmaxf(v, 0.f));
      }
    }
  }
}

// ---------------------------------------------------------------------------
extern "C" void kernel_launch(void* const* d_in, const int* in_sizes, int n_in,
                              void* d_out, int out_size, void* d_ws, size_t ws_size,
                              hipStream_t stream) {
  const float* x      = (const float*)d_in[0];
  const float* Wk     = (const float*)d_in[1];
  const float* Wq     = (const float*)d_in[2];
  const float* Wv     = (const float*)d_in[3];
  const float* proj_w = (const float*)d_in[4];
  const float* proj_b = (const float*)d_in[5];
  const float* ln1_g  = (const float*)d_in[6];
  const float* ln1_b  = (const float*)d_in[7];
  const float* W1     = (const float*)d_in[8];
  const float* b1     = (const float*)d_in[9];
  const float* W2     = (const float*)d_in[10];
  const float* b2     = (const float*)d_in[11];
  const float* ln2_g  = (const float*)d_in[12];
  const float* ln2_b  = (const float*)d_in[13];
  float* out = (float*)d_out;

  char* ws = (char*)d_ws;
  size_t off = 0;
  auto alloc = [&](size_t bytes) -> void* {
    void* p = ws + off; off += (bytes + 255) & ~(size_t)255; return p;
  };
  unsigned short* hln    = (unsigned short*)alloc((size_t)MROWS * CEMB * 2);
  unsigned short* h2ln   = (unsigned short*)alloc((size_t)MROWS * CEMB * 2);
  unsigned short* WqkvT  = (unsigned short*)alloc((size_t)3 * CEMB * CEMB * 2);
  unsigned short* projT  = (unsigned short*)alloc((size_t)CEMB * CEMB * 2);
  unsigned short* W1T    = (unsigned short*)alloc((size_t)FF * CEMB * 2);
  unsigned short* W2T    = (unsigned short*)alloc((size_t)CEMB * FF * 2);
  unsigned short* kbuf   = (unsigned short*)alloc((size_t)BATCH * NHEAD * SEQ * HDIM * 2);
  unsigned short* qbuf   = (unsigned short*)alloc((size_t)BATCH * NHEAD * SEQ * HDIM * 2);
  unsigned short* vTbuf  = (unsigned short*)alloc((size_t)BATCH * NHEAD * HDIM * SEQ * 2);
  unsigned short* attbf  = (unsigned short*)alloc((size_t)MROWS * CEMB * 2);
  unsigned short* hrelu  = (unsigned short*)alloc((size_t)MROWS * FF * 2);
  float*          x1     = (float*)alloc((size_t)MROWS * CEMB * 4);

  // 1) weights -> transposed bf16
  conv_qkvT_kernel<<<(3 * CEMB * CEMB + 255) / 256, 256, 0, stream>>>(Wk, Wq, Wv, WqkvT);
  convT_kernel<<<(CEMB * CEMB + 255) / 256, 256, 0, stream>>>(proj_w, projT, CEMB, CEMB);
  convT_kernel<<<(CEMB * FF + 255) / 256, 256, 0, stream>>>(W1, W1T, CEMB, FF);
  convT_kernel<<<(FF * CEMB + 255) / 256, 256, 0, stream>>>(W2, W2T, FF, CEMB);

  // 2) LN1
  ln_bf16_kernel<<<MROWS, 256, 0, stream>>>(x, ln1_g, ln1_b, hln);

  // 3) QKV projection: 16384 x 1536 x 512  (32x64 per wave -> 1536 blocks)
  gemm_qkv_kernel<<<(MROWS / 32) * ((3 * CEMB) / 64) / 8, 256, 0, stream>>>(
      hln, WqkvT, kbuf, qbuf, vTbuf);

  // 4) attention
  {
    size_t smem = 32 * 256 * 4 + 32 * 256 * 2 + 32 * 8 * 4;   // 50176 B
    attn_kernel<<<BATCH * NHEAD * (SEQ / 32), 256, smem, stream>>>(kbuf, qbuf, vTbuf, attbf);
  }

  // 5) out-proj + residual: x1 = x + att @ proj_w + proj_b
  gemm_bias_kernel<0><<<(MROWS / 32) * (CEMB / 64) / 8, 256, 0, stream>>>(
      attbf, projT, CEMB, CEMB, proj_b, x, x1, nullptr);

  // 6) LN2
  ln_bf16_kernel<<<MROWS, 256, 0, stream>>>(x1, ln2_g, ln2_b, h2ln);

  // 7) MLP1: relu(h2 @ W1 + b1)
  gemm_bias_kernel<1><<<(MROWS / 32) * (FF / 64) / 8, 256, 0, stream>>>(
      h2ln, W1T, FF, CEMB, b1, nullptr, nullptr, hrelu);

  // 8) MLP2 + residual -> d_out
  gemm_bias_kernel<0><<<(MROWS / 32) * (CEMB / 64) / 8, 256, 0, stream>>>(
      hrelu, W2T, CEMB, FF, b2, x1, out, nullptr);

  (void)in_sizes; (void)n_in; (void)out_size; (void)ws_size;
}
